// EncoderLayer_19731079758060
// MI455X (gfx1250) — compile-verified, matrix operand order checked by reading, not run on hardware
//
#include <hip/hip_runtime.h>

// ---------------------------------------------------------------------------
// Types
// ---------------------------------------------------------------------------
typedef __bf16 bf16_t;
typedef bf16_t v16bf __attribute__((ext_vector_type(16)));
typedef bf16_t v8bf  __attribute__((ext_vector_type(8)));
typedef float  v8f   __attribute__((ext_vector_type(8)));

static constexpr int Bc  = 4;
static constexpr int Sc  = 2048;
static constexpr int Dc  = 1024;
static constexpr int Hc  = 16;
static constexpr int DKc = 64;
static constexpr int DFFc = 4096;
static constexpr int Mrows = Bc * Sc;  // 8192

// ---------------------------------------------------------------------------
// Helpers
// ---------------------------------------------------------------------------
__device__ __forceinline__ bf16_t f2bf(float f) {
  unsigned u = __builtin_bit_cast(unsigned, f);
  u = (u + 0x7FFFu + ((u >> 16) & 1u)) >> 16;
  unsigned short s = (unsigned short)u;
  return __builtin_bit_cast(bf16_t, s);
}

__device__ __forceinline__ v8f vzero8() {
  v8f z;
#pragma unroll
  for (int i = 0; i < 8; ++i) z[i] = 0.f;
  return z;
}

// Build a 16-element bf16 A/B-operand fragment from two contiguous 8-elem runs.
__device__ __forceinline__ v16bf make_frag(const bf16_t* lo, const bf16_t* hi) {
  v8bf a = *(const v8bf*)lo;
  v8bf b = *(const v8bf*)hi;
  return __builtin_shufflevector(a, b, 0, 1, 2, 3, 4, 5, 6, 7,
                                 8, 9, 10, 11, 12, 13, 14, 15);
}

__device__ __forceinline__ v8f wmma_bf16(v16bf a, v16bf b, v8f c) {
  return __builtin_amdgcn_wmma_f32_16x16x32_bf16(
      /*neg_a=*/false, a, /*neg_b=*/false, b,
      /*c_mod=*/(short)0, c, /*reuse_a=*/false, /*reuse_b=*/false);
}

// Async global -> LDS copy (16B per lane), tracked by ASYNCcnt.
// LDS aperture keeps the wave-relative LDS offset in addr[31:0], so truncating
// the generic pointer yields the VDST LDS byte address.
__device__ __forceinline__ void async_copy_b128(void* lds, const void* gptr) {
  const unsigned lds_off = (unsigned)(unsigned long long)lds;
  asm volatile("global_load_async_to_lds_b128 %0, %1, off"
               :
               : "v"(lds_off), "v"(gptr)
               : "memory");
}
__device__ __forceinline__ void wait_asynccnt0() {
  asm volatile("s_wait_asynccnt 0x0" ::: "memory");
}

__device__ __forceinline__ float half_reduce_max(float v) {
  v = fmaxf(v, __shfl_xor(v, 1, 32));
  v = fmaxf(v, __shfl_xor(v, 2, 32));
  v = fmaxf(v, __shfl_xor(v, 4, 32));
  v = fmaxf(v, __shfl_xor(v, 8, 32));
  return v;
}
__device__ __forceinline__ float half_reduce_sum(float v) {
  v += __shfl_xor(v, 1, 32);
  v += __shfl_xor(v, 2, 32);
  v += __shfl_xor(v, 4, 32);
  v += __shfl_xor(v, 8, 32);
  return v;
}

__device__ __forceinline__ float block_sum256(float v) {
#pragma unroll
  for (int m = 16; m >= 1; m >>= 1) v += __shfl_xor(v, m, 32);
  __shared__ float sred[8];
  const int w = threadIdx.x >> 5;
  if ((threadIdx.x & 31) == 0) sred[w] = v;
  __syncthreads();
  float r = 0.f;
#pragma unroll
  for (int i = 0; i < 8; ++i) r += sred[i];
  __syncthreads();
  return r;
}

// ---------------------------------------------------------------------------
// Elementwise fp32 -> bf16
// ---------------------------------------------------------------------------
__global__ __launch_bounds__(256) void cvt_kernel(const float* __restrict__ in,
                                                  bf16_t* __restrict__ out) {
  const int i = (blockIdx.x * 256 + threadIdx.x) * 4;
#pragma unroll
  for (int e = 0; e < 4; ++e) out[i + e] = f2bf(in[i + e]);
}

// ---------------------------------------------------------------------------
// Weight transpose+convert: W[K][N] fp32 -> Wt[N][K] bf16
// ---------------------------------------------------------------------------
__global__ __launch_bounds__(256) void wtrans_kernel(const float* __restrict__ W,
                                                     bf16_t* __restrict__ Wt,
                                                     int K, int N) {
  __shared__ float tile[32][33];
  const int k0 = blockIdx.y * 32, n0 = blockIdx.x * 32;
  const int tx = threadIdx.x & 31, ty = threadIdx.x >> 5;
#pragma unroll
  for (int i = ty; i < 32; i += 8)
    tile[i][tx] = W[(size_t)(k0 + i) * N + n0 + tx];
  __syncthreads();
#pragma unroll
  for (int i = ty; i < 32; i += 8)
    Wt[(size_t)(n0 + i) * K + k0 + tx] = f2bf(tile[tx][i]);
}

// ---------------------------------------------------------------------------
// V transpose (bf16): V[B,S,H,DK] -> Vt[B,H,DK,S]
// ---------------------------------------------------------------------------
__global__ __launch_bounds__(256) void vtrans_kernel(const bf16_t* __restrict__ V,
                                                     bf16_t* __restrict__ Vt) {
  __shared__ __align__(16) bf16_t tl[32][72];
  const int b = blockIdx.z, h = blockIdx.y, s0 = blockIdx.x * 32;
  const int t = threadIdx.x;
  {
    const int i = t >> 3, c = (t & 7) * 8;
    *(v8bf*)&tl[i][c] =
        *(const v8bf*)(V + ((size_t)(b * Sc + s0 + i)) * Dc + h * DKc + c);
  }
  __syncthreads();
  {
    const int dk = t >> 2, sc = (t & 3) * 8;
    v8bf o;
#pragma unroll
    for (int e = 0; e < 8; ++e) o[e] = tl[sc + e][dk];
    *(v8bf*)(Vt + (((size_t)(b * Hc + h)) * DKc + dk) * Sc + s0 + sc) = o;
  }
}

// ---------------------------------------------------------------------------
// bf16 WMMA GEMM: C[M,N] = A[M,K] @ Bt[N,K]^T + bias ; optional ReLU.
// 128x128x32 block tile, 256 threads = 8 waves (wave32), 2x4 wmma tiles/wave,
// double-buffered LDS staged with async global->LDS copies (ASYNCcnt).
// ---------------------------------------------------------------------------
#define GBM 128
#define GBN 128
#define GBK 32
#define GLD 40  // LDS row stride in bf16 (80B, 16B aligned)

__global__ __launch_bounds__(256) void gemm_bf16_kernel(
    const bf16_t* __restrict__ A, const bf16_t* __restrict__ Bt,
    const float* __restrict__ bias, float* __restrict__ Cf,
    bf16_t* __restrict__ Cb, int M, int N, int K, int relu) {
  __shared__ __align__(16) bf16_t sA[2][GBM][GLD];
  __shared__ __align__(16) bf16_t sB[2][GBN][GLD];

  const int t = threadIdx.x;
  const int w = t >> 5, lane = t & 31;
  const int r = lane & 15, half = lane >> 4;
  const int wm = w & 3, wn = w >> 2;  // 4x2 wave grid: 32 rows x 64 cols each
  const int bm0 = blockIdx.y * GBM, bn0 = blockIdx.x * GBN;

  v8f acc[2][4];
#pragma unroll
  for (int mt = 0; mt < 2; ++mt)
#pragma unroll
    for (int nt = 0; nt < 4; ++nt) acc[mt][nt] = vzero8();

  const int lrow = t >> 2;         // 0..63
  const int lcol = (t & 3) * 8;    // 0,8,16,24

  auto load_tile = [&](int buf, int k0) {
#pragma unroll
    for (int i = 0; i < 2; ++i) {
      const int row = lrow + i * 64;
      async_copy_b128(&sA[buf][row][lcol],
                      A + (size_t)(bm0 + row) * K + k0 + lcol);
      async_copy_b128(&sB[buf][row][lcol],
                      Bt + (size_t)(bn0 + row) * K + k0 + lcol);
    }
  };

  load_tile(0, 0);
  wait_asynccnt0();
  __syncthreads();

  const int nk = K / GBK;
  for (int kk = 0; kk < nk; ++kk) {
    const int buf = kk & 1;
    if (kk + 1 < nk) load_tile(buf ^ 1, (kk + 1) * GBK);

    v16bf af[2], bfr[4];
#pragma unroll
    for (int mt = 0; mt < 2; ++mt) {
      const bf16_t* arow = &sA[buf][wm * 32 + mt * 16 + r][0];
      af[mt] = make_frag(arow + half * 8, arow + 16 + half * 8);
    }
#pragma unroll
    for (int nt = 0; nt < 4; ++nt) {
      const bf16_t* brow = &sB[buf][wn * 64 + nt * 16 + r][0];
      bfr[nt] = make_frag(brow + half * 8, brow + 16 + half * 8);
    }
#pragma unroll
    for (int mt = 0; mt < 2; ++mt)
#pragma unroll
      for (int nt = 0; nt < 4; ++nt)
        acc[mt][nt] = wmma_bf16(af[mt], bfr[nt], acc[mt][nt]);

    wait_asynccnt0();  // next buffer fully resident in LDS
    __syncthreads();
  }

  // Epilogue: C-layout row = rr + half*8 within tile, col = r within tile
#pragma unroll
  for (int mt = 0; mt < 2; ++mt) {
#pragma unroll
    for (int nt = 0; nt < 4; ++nt) {
      const int col = bn0 + wn * 64 + nt * 16 + r;
      const float bv = bias[col];
#pragma unroll
      for (int rr = 0; rr < 8; ++rr) {
        const int row = bm0 + wm * 32 + mt * 16 + rr + half * 8;
        float v = acc[mt][nt][rr] + bv;
        if (relu) v = fmaxf(v, 0.f);
        if (Cf) Cf[(size_t)row * N + col] = v;
        if (Cb) Cb[(size_t)row * N + col] = f2bf(v);
      }
    }
  }
}

// ---------------------------------------------------------------------------
// Flash attention: per (b,h), 64 Q rows per block (4 waves x 16 rows),
// stream keys in chunks of 32. Q/K bf16 [B,S,D] layout (head-sliced),
// Vt bf16 [B,H,DK,S]. Output ctx bf16 [B,S,D].
// ---------------------------------------------------------------------------
__global__ __launch_bounds__(128) void attn_kernel(
    const bf16_t* __restrict__ Qb, const bf16_t* __restrict__ Kb,
    const bf16_t* __restrict__ Vt, const int* __restrict__ mask,
    bf16_t* __restrict__ ctx) {
  __shared__ __align__(16) bf16_t sP[4][16][GLD];

  const int t = threadIdx.x, w = t >> 5, lane = t & 31;
  const int r = lane & 15, half = lane >> 4;
  const int b = blockIdx.z, h = blockIdx.y;
  const int q0 = blockIdx.x * 64 + w * 16;
  const float scale = 0.125f;  // 1/sqrt(64)

  // Q A-operand fragments held in registers for the whole key loop.
  const bf16_t* qrow = Qb + ((size_t)(b * Sc + q0 + r)) * Dc + h * DKc;
  v16bf qf[2];
  qf[0] = make_frag(qrow + half * 8, qrow + 16 + half * 8);
  qf[1] = make_frag(qrow + 32 + half * 8, qrow + 48 + half * 8);

  v8f acc[4];
#pragma unroll
  for (int nt = 0; nt < 4; ++nt) acc[nt] = vzero8();
  float m[8], l[8];
#pragma unroll
  for (int rr = 0; rr < 8; ++rr) { m[rr] = -1e30f; l[rr] = 0.f; }

  for (int kt = 0; kt < Sc; kt += 32) {
    // --- scores = Q (16x64) @ K_chunk^T (64x32): 2 N-tiles x 2 k-steps ---
    v16bf kb[2][2];
#pragma unroll
    for (int nt = 0; nt < 2; ++nt) {
      const int key = kt + nt * 16 + r;
      const bf16_t* krow = Kb + ((size_t)(b * Sc + key)) * Dc + h * DKc;
      kb[nt][0] = make_frag(krow + half * 8, krow + 16 + half * 8);
      kb[nt][1] = make_frag(krow + 32 + half * 8, krow + 48 + half * 8);
    }
    v8f s0 = vzero8(), s1 = vzero8();
    s0 = wmma_bf16(qf[0], kb[0][0], s0);
    s0 = wmma_bf16(qf[1], kb[0][1], s0);
    s1 = wmma_bf16(qf[0], kb[1][0], s1);
    s1 = wmma_bf16(qf[1], kb[1][1], s1);

    // scale + mask (mask shape [B,1,1,S]; column = r for every acc row)
    const int mv0 = mask[b * Sc + kt + r];
    const int mv1 = mask[b * Sc + kt + 16 + r];
#pragma unroll
    for (int rr = 0; rr < 8; ++rr) {
      s0[rr] = mv0 ? s0[rr] * scale : -1e9f;
      s1[rr] = mv1 ? s1[rr] * scale : -1e9f;
    }

    // online softmax per row (rows live in 16-lane halves)
#pragma unroll
    for (int rr = 0; rr < 8; ++rr) {
      float tm = half_reduce_max(fmaxf(s0[rr], s1[rr]));
      const float nm = fmaxf(m[rr], tm);
      const float corr = __expf(m[rr] - nm);
      const float p0 = __expf(s0[rr] - nm);
      const float p1 = __expf(s1[rr] - nm);
      const float rs = half_reduce_sum(p0 + p1);
      l[rr] = l[rr] * corr + rs;
      m[rr] = nm;
#pragma unroll
      for (int nt = 0; nt < 4; ++nt) acc[nt][rr] *= corr;
      s0[rr] = p0;
      s1[rr] = p1;
    }

    // C-layout -> A-layout for P via per-wave LDS tile (16 rows x 32 keys)
#pragma unroll
    for (int rr = 0; rr < 8; ++rr) {
      const int prow = rr + half * 8;
      sP[w][prow][r] = f2bf(s0[rr]);
      sP[w][prow][16 + r] = f2bf(s1[rr]);
    }
    const bf16_t* pr = &sP[w][r][0];
    const v16bf pa = make_frag(pr + half * 8, pr + 16 + half * 8);

    // --- ctx += P (16x32) @ V_chunk (32x64): 4 N-tiles ---
#pragma unroll
    for (int nt = 0; nt < 4; ++nt) {
      const int dk = nt * 16 + r;
      const bf16_t* vrow =
          Vt + (((size_t)(b * Hc + h)) * DKc + dk) * Sc + kt;
      const v16bf vb = make_frag(vrow + half * 8, vrow + 16 + half * 8);
      acc[nt] = wmma_bf16(pa, vb, acc[nt]);
    }
  }

  // normalize and write ctx (bf16, [B,S,D] head-sliced)
#pragma unroll
  for (int rr = 0; rr < 8; ++rr) {
    const float inv = 1.0f / l[rr];
    const int row = q0 + rr + half * 8;
#pragma unroll
    for (int nt = 0; nt < 4; ++nt) {
      ctx[((size_t)(b * Sc + row)) * Dc + h * DKc + nt * 16 + r] =
          f2bf(acc[nt][rr] * inv);
    }
  }
}

// ---------------------------------------------------------------------------
// LayerNorm over D=1024 with fused residual add; writes f32 and optional bf16
// ---------------------------------------------------------------------------
__global__ __launch_bounds__(256) void ln_kernel(
    const float* __restrict__ xa, const float* __restrict__ xb,
    const float* __restrict__ g, const float* __restrict__ be,
    float* __restrict__ of, bf16_t* __restrict__ ob) {
  const int row = blockIdx.x;
  const size_t base = (size_t)row * Dc;
  const int t = threadIdx.x;
  float v[4];
  float s = 0.f;
#pragma unroll
  for (int e = 0; e < 4; ++e) {
    const int c = e * 256 + t;
    v[e] = xa[base + c] + xb[base + c];
    s += v[e];
  }
  s = block_sum256(s);
  const float mu = s * (1.0f / Dc);
  float vs = 0.f;
#pragma unroll
  for (int e = 0; e < 4; ++e) {
    const float d = v[e] - mu;
    vs += d * d;
  }
  vs = block_sum256(vs);
  const float rstd = rsqrtf(vs * (1.0f / Dc) + 1e-5f);
#pragma unroll
  for (int e = 0; e < 4; ++e) {
    const int c = e * 256 + t;
    const float o = (v[e] - mu) * rstd * g[c] + be[c];
    of[base + c] = o;
    if (ob) ob[base + c] = f2bf(o);
  }
}

// ---------------------------------------------------------------------------
// Host launcher
// ---------------------------------------------------------------------------
extern "C" void kernel_launch(void* const* d_in, const int* in_sizes, int n_in,
                              void* d_out, int out_size, void* d_ws,
                              size_t ws_size, hipStream_t stream) {
  (void)in_sizes; (void)n_in; (void)out_size; (void)ws_size;

  const float* x    = (const float*)d_in[0];
  const int*   mask = (const int*)d_in[1];
  const float* wq = (const float*)d_in[2];   const float* bq = (const float*)d_in[3];
  const float* wk = (const float*)d_in[4];   const float* bk = (const float*)d_in[5];
  const float* wv = (const float*)d_in[6];   const float* bv = (const float*)d_in[7];
  const float* wo = (const float*)d_in[8];   const float* bo = (const float*)d_in[9];
  const float* w1 = (const float*)d_in[10];  const float* b1 = (const float*)d_in[11];
  const float* w2 = (const float*)d_in[12];  const float* b2 = (const float*)d_in[13];
  const float* g1 = (const float*)d_in[14];  const float* be1 = (const float*)d_in[15];
  const float* g2 = (const float*)d_in[16];  const float* be2 = (const float*)d_in[17];

  char* p = (char*)d_ws;
  auto alloc = [&](size_t bytes) -> void* {
    void* r = (void*)p;
    p += (bytes + 255) & ~(size_t)255;
    return r;
  };

  const size_t ND = (size_t)Mrows * Dc;      // 8192*1024
  const size_t NF = (size_t)Mrows * DFFc;    // 8192*4096

  bf16_t* xb    = (bf16_t*)alloc(ND * 2);
  bf16_t* wqT   = (bf16_t*)alloc((size_t)Dc * Dc * 2);
  bf16_t* wkT   = (bf16_t*)alloc((size_t)Dc * Dc * 2);
  bf16_t* wvT   = (bf16_t*)alloc((size_t)Dc * Dc * 2);
  bf16_t* woT   = (bf16_t*)alloc((size_t)Dc * Dc * 2);
  bf16_t* w1T   = (bf16_t*)alloc((size_t)Dc * DFFc * 2);
  bf16_t* w2T   = (bf16_t*)alloc((size_t)DFFc * Dc * 2);
  bf16_t* Qb    = (bf16_t*)alloc(ND * 2);
  bf16_t* Kb    = (bf16_t*)alloc(ND * 2);
  bf16_t* Vb    = (bf16_t*)alloc(ND * 2);
  bf16_t* Vt    = (bf16_t*)alloc(ND * 2);
  bf16_t* ctxb  = (bf16_t*)alloc(ND * 2);
  float*  attnO = (float*)alloc(ND * 4);
  float*  out1  = (float*)alloc(ND * 4);
  bf16_t* out1b = (bf16_t*)alloc(ND * 2);
  bf16_t* h1    = (bf16_t*)alloc(NF * 2);
  float*  ffnO  = (float*)alloc(ND * 4);

  // 1) x -> bf16
  cvt_kernel<<<(int)(ND / 1024), 256, 0, stream>>>(x, xb);

  // 2) weights -> transposed bf16
  wtrans_kernel<<<dim3(Dc / 32, Dc / 32), 256, 0, stream>>>(wq, wqT, Dc, Dc);
  wtrans_kernel<<<dim3(Dc / 32, Dc / 32), 256, 0, stream>>>(wk, wkT, Dc, Dc);
  wtrans_kernel<<<dim3(Dc / 32, Dc / 32), 256, 0, stream>>>(wv, wvT, Dc, Dc);
  wtrans_kernel<<<dim3(Dc / 32, Dc / 32), 256, 0, stream>>>(wo, woT, Dc, Dc);
  wtrans_kernel<<<dim3(DFFc / 32, Dc / 32), 256, 0, stream>>>(w1, w1T, Dc, DFFc);
  wtrans_kernel<<<dim3(Dc / 32, DFFc / 32), 256, 0, stream>>>(w2, w2T, DFFc, Dc);

  // 3) Q/K/V projections (bf16 out)
  gemm_bf16_kernel<<<dim3(Dc / GBN, Mrows / GBM), 256, 0, stream>>>(
      xb, wqT, bq, nullptr, Qb, Mrows, Dc, Dc, 0);
  gemm_bf16_kernel<<<dim3(Dc / GBN, Mrows / GBM), 256, 0, stream>>>(
      xb, wkT, bk, nullptr, Kb, Mrows, Dc, Dc, 0);
  gemm_bf16_kernel<<<dim3(Dc / GBN, Mrows / GBM), 256, 0, stream>>>(
      xb, wvT, bv, nullptr, Vb, Mrows, Dc, Dc, 0);

  // 4) V -> Vt [B,H,DK,S]
  vtrans_kernel<<<dim3(Sc / 32, Hc, Bc), 256, 0, stream>>>(Vb, Vt);

  // 5) flash attention -> ctx (bf16)
  attn_kernel<<<dim3(Sc / 64, Hc, Bc), 128, 0, stream>>>(Qb, Kb, Vt, mask, ctxb);

  // 6) output projection (f32 out)
  gemm_bf16_kernel<<<dim3(Dc / GBN, Mrows / GBM), 256, 0, stream>>>(
      ctxb, woT, bo, attnO, nullptr, Mrows, Dc, Dc, 0);

  // 7) LN1: (x + attn_out) -> out1 f32 + bf16
  ln_kernel<<<Mrows, 256, 0, stream>>>(x, attnO, g1, be1, out1, out1b);

  // 8) FFN1 with ReLU (bf16 out)
  gemm_bf16_kernel<<<dim3(DFFc / GBN, Mrows / GBM), 256, 0, stream>>>(
      out1b, w1T, b1, nullptr, h1, Mrows, DFFc, Dc, 1);

  // 9) FFN2 (f32 out)
  gemm_bf16_kernel<<<dim3(Dc / GBN, Mrows / GBM), 256, 0, stream>>>(
      h1, w2T, b2, ffnO, nullptr, Mrows, Dc, DFFc, 0);

  // 10) LN2: (out1 + ffn) -> d_out
  ln_kernel<<<Mrows, 256, 0, stream>>>(out1, ffnO, g2, be2, (float*)d_out,
                                       nullptr);
}
